// GCN_3040836845699
// MI455X (gfx1250) — compile-verified, matrix operand order checked by reading, not run on hardware
//
#include <hip/hip_runtime.h>

#define N_NODES 100000
#define N_EDGES 3200000
#define N_GRAPHS 512
#define MTILES 6250          // 100000 / 16 exactly
#define KSTRIDE 128          // f16 feature row stride (halves)
#define WSTRIDE 96           // f32 hW/acc row stride (floats)

typedef _Float16 v16h __attribute__((ext_vector_type(16)));
typedef _Float16 v8h  __attribute__((ext_vector_type(8)));
typedef float    v8f  __attribute__((ext_vector_type(8)));

__device__ __forceinline__ void atomAddF(float* p, float v) {
    unsafeAtomicAdd(p, v);   // lowers to global_atomic_add_f32 on gfx1250
}

// ---------------- utility fills ----------------
__global__ void fillf(float* __restrict__ p, float v, size_t n) {
    size_t i = (size_t)blockIdx.x * blockDim.x + threadIdx.x;
    size_t stride = (size_t)gridDim.x * blockDim.x;
    for (; i < n; i += stride) p[i] = v;
}

// ---------------- degree / norm ----------------
__global__ void deg_accum(const int* __restrict__ dst, float* __restrict__ deg) {
    int e = blockIdx.x * blockDim.x + threadIdx.x;
    if (e < N_EDGES) atomAddF(&deg[dst[e]], 1.0f);
}

__global__ void deg_rsqrt(float* __restrict__ d) {
    int i = blockIdx.x * blockDim.x + threadIdx.x;
    if (i < N_NODES) d[i] = rsqrtf(d[i]);   // deg >= 1 (self loop)
}

__global__ void edge_norm(const int* __restrict__ src, const int* __restrict__ dst,
                          const float* __restrict__ dinv, float* __restrict__ nrm) {
    int e = blockIdx.x * blockDim.x + threadIdx.x;
    if (e < N_EDGES) nrm[e] = dinv[src[e]] * dinv[dst[e]];
}

// ---------------- layer 1: relu(x @ W1 + b1) -> f16 padded ----------------
__global__ __launch_bounds__(128) void layer1(const float* __restrict__ x,
                                              const float* __restrict__ W1,
                                              const float* __restrict__ b1,
                                              _Float16* __restrict__ h) {
    __shared__ float xr[19];
    int n = blockIdx.x, t = threadIdx.x;
    if (t < 19) xr[t] = x[n * 19 + t];
    __syncthreads();
    float v = 0.f;
    if (t < 100) {
        v = b1[t];
        #pragma unroll
        for (int k = 0; k < 19; ++k) v += xr[k] * W1[k * 100 + t];
        v = fmaxf(v, 0.f);
    }
    h[(size_t)n * KSTRIDE + t] = (_Float16)v;  // zero padding for t >= 100
}

// ---------------- weight convert: Wt[n][k] = W[k][n] (f16, padded) ----------------
__global__ __launch_bounds__(128) void wconv(const float* __restrict__ W,
                                             _Float16* __restrict__ Wt,
                                             int indim, int outdim) {
    int n = blockIdx.x, k = threadIdx.x;
    float v = (k < indim && n < outdim) ? W[k * outdim + n] : 0.f;
    Wt[(size_t)n * KSTRIDE + k] = (_Float16)v;
}

// ---------------- WMMA GEMM: hW[100000, <=96] = hIn @ W ----------------
// One wave per 16x16 output tile. A: lane holds row m=lane&15, two 16B chunks
// per K-tile (K = j<8?base : base+16, with +8 for lanes 16..31). B = W^T rows
// (column-major W), same striping. C/D per 8-VGPR f32 layout.
__global__ __launch_bounds__(32) void wmma_gemm(const _Float16* __restrict__ hIn,
                                                const _Float16* __restrict__ Wt,
                                                float* __restrict__ hW, int ktiles) {
    int lane = threadIdx.x;
    int tm = blockIdx.x, tn = blockIdx.y;
    int koff = (lane >> 4) * 8;
    const _Float16* pa = hIn + (size_t)(tm * 16 + (lane & 15)) * KSTRIDE + koff;
    const _Float16* pb = Wt  + (size_t)(tn * 16 + (lane & 15)) * KSTRIDE + koff;
    v8f c = {};
    for (int kt = 0; kt < ktiles; ++kt) {
        v8h a0 = *(const v8h*)(pa);
        v8h a1 = *(const v8h*)(pa + 16);
        v8h b0 = *(const v8h*)(pb);
        v8h b1 = *(const v8h*)(pb + 16);
        v16h a = __builtin_shufflevector(a0, a1, 0,1,2,3,4,5,6,7,8,9,10,11,12,13,14,15);
        v16h b = __builtin_shufflevector(b0, b1, 0,1,2,3,4,5,6,7,8,9,10,11,12,13,14,15);
        c = __builtin_amdgcn_wmma_f32_16x16x32_f16(false, a, false, b,
                                                   (short)0, c, false, false);
        pa += 32; pb += 32;
    }
    int row = tm * 16 + (lane >> 4) * 8;
    int col = tn * 16 + (lane & 15);
    #pragma unroll
    for (int v = 0; v < 8; ++v)
        hW[(size_t)(row + v) * WSTRIDE + col] = c[v];
}

// ---------------- edge scatter: acc[dst] += hW[src] * norm ----------------
__global__ __launch_bounds__(96) void scatter(const int* __restrict__ src,
                                              const int* __restrict__ dst,
                                              const float* __restrict__ nrm,
                                              const float* __restrict__ hW,
                                              float* __restrict__ acc, int outdim) {
    int e = blockIdx.x, c = threadIdx.x;
    if (c < outdim) {
        int s = src[e], d = dst[e];
        atomAddF(&acc[(size_t)d * WSTRIDE + c], hW[(size_t)s * WSTRIDE + c] * nrm[e]);
    }
}

// ---------------- finalize conv: relu(acc + selfloop + b) -> f16 padded ----------------
__global__ __launch_bounds__(128) void finalize(const float* __restrict__ acc,
                                                const float* __restrict__ hW,
                                                const float* __restrict__ dinv,
                                                const float* __restrict__ b,
                                                _Float16* __restrict__ hNext, int outdim) {
    int n = blockIdx.x, t = threadIdx.x;
    float r = 0.f;
    if (t < outdim) {
        float di = dinv[n];
        r = fmaxf(acc[(size_t)n * WSTRIDE + t] + hW[(size_t)n * WSTRIDE + t] * di * di + b[t], 0.f);
    }
    hNext[(size_t)n * KSTRIDE + t] = (_Float16)r;
}

// ---------------- segment max (relu'd values >= 0 -> uint-bit monotone) ----------------
__global__ __launch_bounds__(64) void segmax(const _Float16* __restrict__ h,
                                             const int* __restrict__ batch,
                                             float* __restrict__ g) {
    int n = blockIdx.x, t = threadIdx.x;
    if (t < 50) {
        float v = (float)h[(size_t)n * KSTRIDE + t];
        atomicMax((unsigned int*)&g[(size_t)batch[n] * 64 + t], __float_as_uint(v));
    }
}

// ---------------- MLP head + softmax over graph axis ----------------
__global__ __launch_bounds__(512) void mlp_softmax(const float* __restrict__ g,
    const float* __restrict__ W2, const float* __restrict__ b2,
    const float* __restrict__ W3, const float* __restrict__ b3,
    const float* __restrict__ W4, const float* __restrict__ b4,
    const float* __restrict__ W5, const float* __restrict__ b5,
    const float* __restrict__ W6, const float* __restrict__ b6,
    float* __restrict__ out) {
    __shared__ float lg[N_GRAPHS * 6];
    __shared__ float mx[6], sm[6];
    int r = threadIdx.x;
    float a[50], t[40];
    #pragma unroll
    for (int i = 0; i < 50; ++i) a[i] = g[(size_t)r * 64 + i];
    for (int o = 0; o < 40; ++o) { float s = b2[o]; for (int k = 0; k < 50; ++k) s += a[k] * W2[k * 40 + o]; t[o] = fmaxf(s, 0.f); }
    for (int o = 0; o < 30; ++o) { float s = b3[o]; for (int k = 0; k < 40; ++k) s += t[k] * W3[k * 30 + o]; a[o] = fmaxf(s, 0.f); }
    for (int o = 0; o < 20; ++o) { float s = b4[o]; for (int k = 0; k < 30; ++k) s += a[k] * W4[k * 20 + o]; t[o] = fmaxf(s, 0.f); }
    for (int o = 0; o < 10; ++o) { float s = b5[o]; for (int k = 0; k < 20; ++k) s += t[k] * W5[k * 10 + o]; a[o] = fmaxf(s, 0.f); }
    for (int o = 0; o <  6; ++o) { float s = b6[o]; for (int k = 0; k < 10; ++k) s += a[k] * W6[k * 6 + o]; lg[r * 6 + o] = s; }
    __syncthreads();
    if (r < 6) {
        float m = -3.4e38f;
        for (int i = 0; i < N_GRAPHS; ++i) m = fmaxf(m, lg[i * 6 + r]);
        float s = 0.f;
        for (int i = 0; i < N_GRAPHS; ++i) s += __expf(lg[i * 6 + r] - m);
        mx[r] = m; sm[r] = s;
    }
    __syncthreads();
    for (int c = 0; c < 6; ++c)
        out[r * 6 + c] = __expf(lg[r * 6 + c] - mx[c]) / sm[c];
}

extern "C" void kernel_launch(void* const* d_in, const int* in_sizes, int n_in,
                              void* d_out, int out_size, void* d_ws, size_t ws_size,
                              hipStream_t stream) {
    const float* x    = (const float*)d_in[0];
    const int*   ei   = (const int*)d_in[1];
    const int*   srcI = ei;
    const int*   dstI = ei + N_EDGES;
    const int*   batch = (const int*)d_in[2];
    const float* W1  = (const float*)d_in[3];  const float* b1  = (const float*)d_in[4];
    const float* Wc1 = (const float*)d_in[5];  const float* bc1 = (const float*)d_in[6];
    const float* Wc2 = (const float*)d_in[7];  const float* bc2 = (const float*)d_in[8];
    const float* Wc3 = (const float*)d_in[9];  const float* bc3 = (const float*)d_in[10];
    const float* W2  = (const float*)d_in[11]; const float* b2  = (const float*)d_in[12];
    const float* W3  = (const float*)d_in[13]; const float* b3  = (const float*)d_in[14];
    const float* W4  = (const float*)d_in[15]; const float* b4  = (const float*)d_in[16];
    const float* W5  = (const float*)d_in[17]; const float* b5  = (const float*)d_in[18];
    const float* W6  = (const float*)d_in[19]; const float* b6  = (const float*)d_in[20];

    char* ws = (char*)d_ws;
    size_t off = 0;
    auto alloc = [&](size_t bytes) { size_t r = off; off = (off + bytes + 255) & ~(size_t)255; return r; };
    float*    dinv = (float*)   (ws + alloc((size_t)N_NODES * 4));
    float*    nrm  = (float*)   (ws + alloc((size_t)N_EDGES * 4));
    _Float16* hA   = (_Float16*)(ws + alloc((size_t)N_NODES * KSTRIDE * 2));
    _Float16* hB   = (_Float16*)(ws + alloc((size_t)N_NODES * KSTRIDE * 2));
    float*    hW   = (float*)   (ws + alloc((size_t)N_NODES * WSTRIDE * 4));
    float*    acc  = (float*)   (ws + alloc((size_t)N_NODES * WSTRIDE * 4));
    _Float16* Wt1  = (_Float16*)(ws + alloc((size_t)96 * KSTRIDE * 2));
    _Float16* Wt2  = (_Float16*)(ws + alloc((size_t)96 * KSTRIDE * 2));
    _Float16* Wt3  = (_Float16*)(ws + alloc((size_t)96 * KSTRIDE * 2));
    float*    g    = (float*)   (ws + alloc((size_t)N_GRAPHS * 64 * 4));
    (void)ws_size; (void)in_sizes; (void)n_in; (void)out_size;

    const int EB = (N_EDGES + 255) / 256;
    const int NB = (N_NODES + 255) / 256;

    // weights -> f16 transposed/padded
    wconv<<<96, 128, 0, stream>>>(Wc1, Wt1, 100, 90);
    wconv<<<80, 128, 0, stream>>>(Wc2, Wt2,  90, 70);
    wconv<<<64, 128, 0, stream>>>(Wc3, Wt3,  70, 50);

    // gcn_norm
    fillf<<<1024, 256, 0, stream>>>(dinv, 1.0f, (size_t)N_NODES);   // self-loop degree
    deg_accum<<<EB, 256, 0, stream>>>(dstI, dinv);
    deg_rsqrt<<<NB, 256, 0, stream>>>(dinv);
    edge_norm<<<EB, 256, 0, stream>>>(srcI, dstI, dinv, nrm);

    // input MLP
    layer1<<<N_NODES, 128, 0, stream>>>(x, W1, b1, hA);

    // conv1: 100 -> 90
    fillf<<<4096, 256, 0, stream>>>(acc, 0.f, (size_t)N_NODES * WSTRIDE);
    wmma_gemm<<<dim3(MTILES, 6), 32, 0, stream>>>(hA, Wt1, hW, 4);
    scatter<<<N_EDGES, 96, 0, stream>>>(srcI, dstI, nrm, hW, acc, 90);
    finalize<<<N_NODES, 128, 0, stream>>>(acc, hW, dinv, bc1, hB, 90);

    // conv2: 90 -> 70
    fillf<<<4096, 256, 0, stream>>>(acc, 0.f, (size_t)N_NODES * WSTRIDE);
    wmma_gemm<<<dim3(MTILES, 5), 32, 0, stream>>>(hB, Wt2, hW, 3);
    scatter<<<N_EDGES, 96, 0, stream>>>(srcI, dstI, nrm, hW, acc, 70);
    finalize<<<N_NODES, 128, 0, stream>>>(acc, hW, dinv, bc2, hA, 70);

    // conv3: 70 -> 50
    fillf<<<4096, 256, 0, stream>>>(acc, 0.f, (size_t)N_NODES * WSTRIDE);
    wmma_gemm<<<dim3(MTILES, 4), 32, 0, stream>>>(hA, Wt3, hW, 3);
    scatter<<<N_EDGES, 96, 0, stream>>>(srcI, dstI, nrm, hW, acc, 50);
    finalize<<<N_NODES, 128, 0, stream>>>(acc, hW, dinv, bc3, hB, 50);

    // per-graph max pooling (values are relu'd -> non-negative, uint-bit max is exact)
    fillf<<<128, 256, 0, stream>>>(g, 0.f, (size_t)N_GRAPHS * 64);
    segmax<<<N_NODES, 64, 0, stream>>>(hB, batch, g);

    // head + softmax over graph axis
    mlp_softmax<<<1, 512, 0, stream>>>(g, W2, b2, W3, b3, W4, b4, W5, b5, W6, b6,
                                       (float*)d_out);
}
